// hyperConv_30459908063658
// MI455X (gfx1250) — matrix operand order repaired and checked by reading, not compiled, not run on hardware
//
#include <hip/hip_runtime.h>

typedef __attribute__((ext_vector_type(16))) __bf16    v16bf;
typedef __attribute__((ext_vector_type(8)))  float     v8f;
typedef __attribute__((ext_vector_type(4)))  unsigned  u32x4;

#define Bn    8
#define CIN   64
#define COUT  64
#define Hh    256
#define Ww    256
#define SD    64
#define WD    8
#define HID   32          // 4*WD

// ---- workspace layout (bytes) ----
// [0..255]      f coefficients (64 f32)
// [512..2559]   bias (512 f32)  (f_buf+128 floats)
// [2560.. ]     bf16 weights  8 * 9*64*64 ushort = 589824 B
#define WS_WBF_BYTE 2560

// ---- LDS geometry ----
#define XC  66            // 64-pixel strip + 2 halo cols
#define IP  80            // padded channel stride (ushort) for x tile  (40 dwords, 16B-aligned rows)
#define WP  80            // padded channel stride (ushort) for weights
#define WBYTES (9*64*WP*2)        // 92160
#define XBYTES (3*XC*IP*2)        // 31680
#define ROWS 8

static __device__ __forceinline__ unsigned short f2bf(float f) {
  unsigned u = __builtin_bit_cast(unsigned, f);
  u += 0x7FFFu + ((u >> 16) & 1u);           // round-to-nearest-even
  return (unsigned short)(u >> 16);
}

// ---------------------------------------------------------------------------
// Kernel 1: hypernet heads  f = (s@W1+b1)@W2+b2  (and fb), bias = b_param @ fb^T
// ---------------------------------------------------------------------------
__global__ void hyper_head(const float* __restrict__ s,
                           const float* __restrict__ fc1_w, const float* __restrict__ fc1_b,
                           const float* __restrict__ fc2_w, const float* __restrict__ fc2_b,
                           const float* __restrict__ fcb1_w, const float* __restrict__ fcb1_b,
                           const float* __restrict__ fcb2_w, const float* __restrict__ fcb2_b,
                           const float* __restrict__ b_param,
                           float* __restrict__ f_out, float* __restrict__ bias_out) {
  __shared__ float fbsh[64];
  const int t = threadIdx.x;
  if (t < 128) {
    const int which = t >> 6;                  // 0 -> f, 1 -> fb
    const int bd = t & 63;
    const int b = bd >> 3, d = bd & 7;
    const float* w1 = which ? fcb1_w : fc1_w;
    const float* b1 = which ? fcb1_b : fc1_b;
    const float* w2 = which ? fcb2_w : fc2_w;
    const float* b2 = which ? fcb2_b : fc2_b;
    float acc = b2[d];
    for (int j = 0; j < HID; ++j) {
      float h = b1[j];
      for (int k = 0; k < SD; ++k) h += s[b*SD + k] * w1[k*HID + j];
      acc += h * w2[j*WD + d];
    }
    if (which == 0) f_out[bd] = acc; else fbsh[bd] = acc;
  }
  __syncthreads();
  const int b = t >> 6, o = t & 63;            // 512 threads -> bias[b][o]
  float a = 0.f;
  #pragma unroll
  for (int d = 0; d < WD; ++d) a += b_param[o*WD + d] * fbsh[(b<<3) + d];
  bias_out[t] = a;
}

// ---------------------------------------------------------------------------
// Kernel 2: materialize per-batch conv kernels as bf16, layout [b][kh*3+kw][o][i]
// ---------------------------------------------------------------------------
__global__ void wker_build(const float* __restrict__ param, const float* __restrict__ f,
                           unsigned short* __restrict__ wbf) {
  const int u = blockIdx.x * 256 + threadIdx.x;          // < 8*36864
  if (u >= Bn * 36864) return;
  const int b   = u / 36864;
  const int rem = u - b * 36864;
  const int t9  = rem >> 12;                             // kh*3+kw
  const int oi  = rem & 4095;                            // o*64+i
  const float* p  = param + (size_t)(oi * 9 + t9) * WD;  // param[o][i][kh][kw][d]
  const float* fb = f + b * WD;
  float acc = 0.f;
  #pragma unroll
  for (int d = 0; d < WD; ++d) acc += p[d] * fb[d];
  wbf[u] = f2bf(acc);
}

// ---------------------------------------------------------------------------
// Kernel 3: implicit-GEMM conv with bf16 WMMA.
// Grid (W/64, H/8, B); 256 threads = 8 waves; wave w -> Mtile w>>1, Ntile pair w&1.
// ---------------------------------------------------------------------------
__global__ void __launch_bounds__(256, 2)
hyperconv_main(const float* __restrict__ x, const unsigned short* __restrict__ wbf,
               const float* __restrict__ biasbuf, float* __restrict__ out) {
  extern __shared__ char smem[];
  unsigned short* wst = (unsigned short*)smem;
  unsigned short* xst = (unsigned short*)(smem + WBYTES);
  unsigned* wsu = (unsigned*)wst;
  unsigned* xsu = (unsigned*)xst;

  const int tid = threadIdx.x;
  const int b   = blockIdx.z;
  const int y0  = blockIdx.y * ROWS;
  const int x0  = blockIdx.x * 64;

  // ---- stage this batch's weights: wbf[b][t9][o][i] -> LDS [t9][o][i(pad 80)] ----
  {
    const unsigned* wsrc = (const unsigned*)(wbf + (size_t)b * 36864);
    for (int p = tid; p < 18432; p += 256) {             // dword pairs, i-contiguous
      const int t9  = p >> 11;
      const int rem = p & 2047;
      const int o   = rem >> 5;
      const int ip  = rem & 31;
      wsu[(t9*64 + o)*(WP/2) + ip] = wsrc[p];
    }
  }

  // rolling 3-row halo window, bf16, channel-innermost
  auto stage_row = [&](int gy) {
    const int slot = (gy + 3) % 3;
    const float* xb = x + (size_t)b * CIN * Hh * Ww;
    for (int idx = tid; idx < CIN * XC; idx += 256) {
      const int i   = idx / XC;
      const int col = idx - i * XC;
      const int gx  = x0 + col - 1;
      float v = 0.f;
      if ((unsigned)gy < (unsigned)Hh && (unsigned)gx < (unsigned)Ww) {
        const float* ptr = xb + ((size_t)i * Hh + gy) * Ww + gx;
        v = *ptr;
        if (gy + 1 < Hh) __builtin_prefetch(ptr + Ww, 0, 0);   // global_prefetch_b8
      }
      xst[(slot*XC + col)*IP + i] = f2bf(v);
    }
  };

  stage_row(y0 - 1);
  stage_row(y0);

  const int lane  = tid & 31;
  const int wave  = tid >> 5;
  const int half  = lane >> 4;
  const int l16   = lane & 15;
  const int mbase = (wave >> 1) * 16;     // 0,16,32,48
  const int nb0   = (wave & 1) * 32;      // N tiles at nb0 and nb0+16

  // per-lane bias for the 8 C-matrix rows (VGPR v -> M = mbase + half*8 + v)
  float bval[8];
  #pragma unroll
  for (int v = 0; v < 8; ++v) bval[v] = biasbuf[b*64 + mbase + half*8 + v];

  for (int y = y0; y < y0 + ROWS; ++y) {
    stage_row(y + 1);
    __syncthreads();

    v8f acc0 = {}, acc1 = {};
    #pragma unroll
    for (int kh = 0; kh < 3; ++kh) {
      const int slot = (y + kh + 2) % 3;                 // row y+kh-1
      #pragma unroll
      for (int kw = 0; kw < 3; ++kw) {
        const int t9 = kh*3 + kw;
        const int abase = (t9*64 + mbase + l16) * (WP/2);
        const int bbase0 = (slot*XC + nb0 + l16 + kw) * (IP/2);
        const int bbase1 = bbase0 + 16 * (IP/2);
        #pragma unroll
        for (int s2 = 0; s2 < 2; ++s2) {                 // channel halves (K=32 each)
          union { v16bf v; u32x4 q[2]; } a, b0f, b1f;
          // A 16x32 bf16: lanes0-15 K0-15, lanes16-31 K16-31
          const int ab = abase + s2*16 + half*8;
          a.q[0] = *(const u32x4*)(wsu + ab);
          a.q[1] = *(const u32x4*)(wsu + ab + 4);
          // B 32x16 bf16: lanes0-15 K0-15 (N=lane), lanes16-31 K16-31
          const int bb = s2*16 + half*8;
          b0f.q[0] = *(const u32x4*)(xsu + bbase0 + bb);
          b0f.q[1] = *(const u32x4*)(xsu + bbase0 + bb + 4);
          b1f.q[0] = *(const u32x4*)(xsu + bbase1 + bb);
          b1f.q[1] = *(const u32x4*)(xsu + bbase1 + bb + 4);
          acc0 = __builtin_amdgcn_wmma_f32_16x16x32_bf16(false, a.v, false, b0f.v,
                                                         (short)0, acc0, false, false);
          acc1 = __builtin_amdgcn_wmma_f32_16x16x32_bf16(false, a.v, false, b1f.v,
                                                         (short)0, acc1, false, false);
        }
      }
    }

    // store row y: C layout VGPR v -> M = mbase+half*8+v, N = nb0+l16 (+16)
    {
      const int n0 = nb0 + l16;
      float* ob = out + ((size_t)(b*64 + mbase + half*8) * Hh + y) * Ww + x0;
      #pragma unroll
      for (int v = 0; v < 8; ++v) {
        float* orow = ob + (size_t)v * Hh * Ww;
        orow[n0]      = acc0[v] + bval[v];
        orow[n0 + 16] = acc1[v] + bval[v];
      }
    }
    __syncthreads();   // protect slot (y-1)%3 before next stage overwrites it
  }
}

// ---------------------------------------------------------------------------
extern "C" void kernel_launch(void* const* d_in, const int* in_sizes, int n_in,
                              void* d_out, int out_size, void* d_ws, size_t ws_size,
                              hipStream_t stream) {
  (void)in_sizes; (void)n_in; (void)out_size; (void)ws_size;
  const float* x       = (const float*)d_in[0];
  const float* s       = (const float*)d_in[1];
  const float* fc1_w   = (const float*)d_in[2];
  const float* fc1_b   = (const float*)d_in[3];
  const float* fc2_w   = (const float*)d_in[4];
  const float* fc2_b   = (const float*)d_in[5];
  const float* fcb1_w  = (const float*)d_in[6];
  const float* fcb1_b  = (const float*)d_in[7];
  const float* fcb2_w  = (const float*)d_in[8];
  const float* fcb2_b  = (const float*)d_in[9];
  const float* param   = (const float*)d_in[10];
  const float* b_param = (const float*)d_in[11];
  float* out = (float*)d_out;

  float* f_buf    = (float*)d_ws;                 // 64 f32
  float* bias_buf = f_buf + 128;                  // 512 f32
  unsigned short* wbf = (unsigned short*)((char*)d_ws + WS_WBF_BYTE);

  hyper_head<<<1, 512, 0, stream>>>(s, fc1_w, fc1_b, fc2_w, fc2_b,
                                    fcb1_w, fcb1_b, fcb2_w, fcb2_b,
                                    b_param, f_buf, bias_buf);
  wker_build<<<(Bn*36864 + 255)/256, 256, 0, stream>>>(param, f_buf, wbf);

  const size_t shmem = (size_t)WBYTES + XBYTES;   // 123840 B dynamic LDS
  (void)hipFuncSetAttribute(reinterpret_cast<const void*>(hyperconv_main),
                            hipFuncAttributeMaxDynamicSharedMemorySize, (int)shmem);
  dim3 grid(Ww/64, Hh/ROWS, Bn);
  hyperconv_main<<<grid, 256, shmem, stream>>>(x, wbf, bias_buf, out);
}
